// CausalSelfAttention_76355928589000
// MI455X (gfx1250) — matrix operand order repaired and checked by reading, not compiled, not run on hardware
//
#include <hip/hip_runtime.h>

// ---------------------------------------------------------------------------
// CDNA5 (gfx1250) causal self-attention.
// ~550 GFLOP vs ~200MB HBM @23.3TB/s -> matrix-core bound. All GEMMs run on
// v_wmma_f32_16x16x32_f16 (wave32). Data movement uses the gfx1250 async path:
// global_load_async_to_lds_b128 + s_wait_asynccnt, double-buffered LDS tiles
// shared by all 8 waves of the workgroup (overlaps copy with WMMA issue).
// ---------------------------------------------------------------------------

typedef __attribute__((ext_vector_type(16))) _Float16 v16h;
typedef __attribute__((ext_vector_type(8)))  _Float16 v8h;
typedef __attribute__((ext_vector_type(4)))  _Float16 v4h;
typedef __attribute__((ext_vector_type(8)))  float    v8f;

union AFrag {
    v16h v;
    struct { v8h lo; v8h hi; } p;
};

// One 16-byte async global->LDS copy (ASYNCcnt-tracked, CDNA5 §10.7/§15.18).
__device__ __forceinline__ void async_copy16(const _Float16* g, _Float16* l) {
    unsigned lds_off = (unsigned)(uintptr_t)l;          // low 32 bits = LDS addr
    unsigned long long ga = (unsigned long long)(uintptr_t)g;
    asm volatile("global_load_async_to_lds_b128 %0, %1, off"
                 :: "v"(lds_off), "v"(ga) : "memory");
}
__device__ __forceinline__ void async_wait0() {
    asm volatile("s_wait_asynccnt 0x0" ::: "memory");
}

// ------------------------- f32 -> f16 cast ---------------------------------
__global__ __launch_bounds__(256) void cast_f32_f16(const float* __restrict__ src,
                                                    _Float16* __restrict__ dst,
                                                    int n) {
    int i = (blockIdx.x * 256 + threadIdx.x) * 4;
    if (i + 3 < n) {
        float4 v = *(const float4*)(src + i);
        v4h o;
        o.x = (_Float16)v.x; o.y = (_Float16)v.y;
        o.z = (_Float16)v.z; o.w = (_Float16)v.w;
        *(v4h*)(dst + i) = o;
    }
}

// ------------------------- V transpose: [B,T,H,d] -> [B,H,d,T] -------------
__global__ __launch_bounds__(256) void transpose_v(const _Float16* __restrict__ qkv,
                                                   _Float16* __restrict__ vT) {
    size_t o = (size_t)blockIdx.x * 256 + threadIdx.x;   // over [B*H*128, T]
    int t    = (int)(o & 2047);
    size_t r = o >> 11;
    int dim  = (int)(r & 127);
    int bh   = (int)(r >> 7);
    int b = bh >> 4, h = bh & 15;
    vT[o] = qkv[(size_t)(b * 2048 + t) * 6144 + 4096 + h * 128 + dim];
}

// ------------------------- WMMA GEMM: C = A[M,K] * W[N,K]^T ----------------
// 256 thr = 8 waves (2x4); block tile 128x128; wave tile 64x32.
// A/B k-slabs (128x32 f16 = 8KB each) double-buffered in LDS via async copies.
template <typename OutT>
__global__ __launch_bounds__(256) void gemm_wmma(const _Float16* __restrict__ A,
                                                 const _Float16* __restrict__ W,
                                                 OutT* __restrict__ Cp,
                                                 int M, int N, int K) {
    __shared__ _Float16 ldsA[2][128 * 32];
    __shared__ _Float16 ldsB[2][128 * 32];

    const int tid  = threadIdx.x;
    const int lane = tid & 31;
    const int wave = tid >> 5;
    const int wm = wave >> 2;            // 0..1
    const int wn = wave & 3;             // 0..3
    const int lh = lane & 15;
    const int lg = lane >> 4;
    const int mBlk = blockIdx.y * 128;
    const int nBlk = blockIdx.x * 128;

    v8f acc[4][2];
#pragma unroll
    for (int i = 0; i < 4; ++i)
#pragma unroll
        for (int j = 0; j < 2; ++j)
            acc[i][j] = (v8f){0.f,0.f,0.f,0.f,0.f,0.f,0.f,0.f};

    // Stage one 128x32 slab of A and of W into LDS buffer `buf`.
    auto stage = [&](int buf, int k) {
#pragma unroll
        for (int j = 0; j < 2; ++j) {
            int c = tid + j * 256;                 // 512 x 16B chunks per tile
            int row = c >> 2, piece = c & 3;       // 4 chunks per 32-half row
            async_copy16(A + (size_t)(mBlk + row) * K + k + piece * 8,
                         &ldsA[buf][c * 8]);
            async_copy16(W + (size_t)(nBlk + row) * K + k + piece * 8,
                         &ldsB[buf][c * 8]);
        }
    };

    stage(0, 0);
    for (int k = 0, it = 0; k < K; k += 32, ++it) {
        const int cur = it & 1;
        async_wait0();          // my async copies for `cur` complete
        __syncthreads();        // everyone's copies done; prev buffer consumed
        if (k + 32 < K) stage(cur ^ 1, k + 32);   // overlap next copy w/ WMMA

        v16h a[4], bf[2];
#pragma unroll
        for (int i = 0; i < 4; ++i) {
            const _Float16* p = &ldsA[cur][(wm * 64 + i * 16 + lh) * 32 + lg * 8];
            AFrag f;
            f.p.lo = *(const v8h*)(p);
            f.p.hi = *(const v8h*)(p + 16);
            a[i] = f.v;
        }
#pragma unroll
        for (int j = 0; j < 2; ++j) {
            const _Float16* q = &ldsB[cur][(wn * 32 + j * 16 + lh) * 32 + lg * 8];
            AFrag g;
            g.p.lo = *(const v8h*)(q);
            g.p.hi = *(const v8h*)(q + 16);
            bf[j] = g.v;
        }

#pragma unroll
        for (int i = 0; i < 4; ++i)
#pragma unroll
            for (int j = 0; j < 2; ++j)
                acc[i][j] = __builtin_amdgcn_wmma_f32_16x16x32_f16(
                    false, a[i], false, bf[j], (short)0, acc[i][j], false, false);
    }

#pragma unroll
    for (int i = 0; i < 4; ++i)
#pragma unroll
        for (int j = 0; j < 2; ++j)
#pragma unroll
            for (int r = 0; r < 8; ++r) {
                int m = mBlk + wm * 64 + i * 16 + lg * 8 + r;
                int n = nBlk + wn * 32 + j * 16 + lh;
                Cp[(size_t)m * N + n] = (OutT)acc[i][j][r];
            }
}

// ------------------------- Flash attention ---------------------------------
// One (b,h) head, 128 q rows per block; wave w owns q rows [w*16, w*16+16).
// K tile (32x128) and V^T tile (128x32) double-buffered in LDS via async
// copies, shared by all 8 waves. Block-uniform kb loop with barriers.
__global__ __launch_bounds__(256) void flash_attn(const _Float16* __restrict__ qkv,
                                                  const _Float16* __restrict__ vT,
                                                  _Float16* __restrict__ out) {
    constexpr int T = 2048, Dh = 128, C = 2048, C3 = 6144;
    __shared__ _Float16 ldsK[2][32 * 128];    // [key][dim]
    __shared__ _Float16 ldsV[2][128 * 32];    // [dim][key]
    __shared__ _Float16 plds[8][16 * 32];     // per-wave P staging

    const int tid  = threadIdx.x;
    const int lane = tid & 31;
    const int wave = tid >> 5;
    const int lh = lane & 15;
    const int lg = lane >> 4;
    const int bh = blockIdx.y;
    const int b = bh >> 4, h = bh & 15;
    const int q0 = blockIdx.x * 128 + wave * 16;
    const float scale = 0.08838834764831845f;        // 1/sqrt(128)

    // Q fragments: 16 rows x 128 dims, resident in VGPRs
    v16h qa[4];
    {
        const _Float16* qp = qkv + (size_t)(b * T + q0 + lh) * C3 + h * Dh + lg * 8;
#pragma unroll
        for (int kk = 0; kk < 4; ++kk) {
            AFrag f;
            f.p.lo = *(const v8h*)(qp + kk * 32);
            f.p.hi = *(const v8h*)(qp + kk * 32 + 16);
            qa[kk] = f.v;
        }
    }

    v8f o[8];
#pragma unroll
    for (int f = 0; f < 8; ++f) o[f] = (v8f){0.f,0.f,0.f,0.f,0.f,0.f,0.f,0.f};
    float rowm[8], rowl[8];
#pragma unroll
    for (int r = 0; r < 8; ++r) { rowm[r] = -1e30f; rowl[r] = 0.f; }

    const int nkbMine = (q0 + 15) / 32 + 1;          // causal bound (this wave)
    const int nkbMax  = blockIdx.x * 4 + 4;          // block-uniform bound

    auto stageKV = [&](int buf, int kb) {
        const int key0 = kb * 32;
#pragma unroll
        for (int j = 0; j < 2; ++j) {
            int c = tid + j * 256;                   // 512 x 16B chunks each
            int key = c >> 4, kp = c & 15;           // K: 16 chunks / 256B row
            async_copy16(qkv + (size_t)(b * T + key0 + key) * C3 + C + h * Dh + kp * 8,
                         &ldsK[buf][c * 8]);
            int dim = c >> 2, vp = c & 3;            // V: 4 chunks / 64B row
            async_copy16(vT + ((size_t)bh * Dh + dim) * T + key0 + vp * 8,
                         &ldsV[buf][c * 8]);
        }
    };

    stageKV(0, 0);
    for (int kb = 0; kb < nkbMax; ++kb) {
        const int cur = kb & 1;
        async_wait0();
        __syncthreads();
        if (kb + 1 < nkbMax) stageKV(cur ^ 1, kb + 1);
        if (kb >= nkbMine) continue;                 // still barriers each iter
        const int key0 = kb * 32;

        // ---- S = Q K^T for 16 q rows x 32 keys (from LDS) ---------------
        v8f s[2];
        s[0] = (v8f){0.f,0.f,0.f,0.f,0.f,0.f,0.f,0.f};
        s[1] = (v8f){0.f,0.f,0.f,0.f,0.f,0.f,0.f,0.f};
#pragma unroll
        for (int nt = 0; nt < 2; ++nt)
#pragma unroll
            for (int kk = 0; kk < 4; ++kk) {
                const _Float16* kp = &ldsK[cur][(nt * 16 + lh) * 128 + kk * 32 + lg * 8];
                AFrag g;
                g.p.lo = *(const v8h*)(kp);
                g.p.hi = *(const v8h*)(kp + 16);
                v16h kf = g.v;
                s[nt] = __builtin_amdgcn_wmma_f32_16x16x32_f16(
                    false, qa[kk], false, kf, (short)0, s[nt], false, false);
            }

        // ---- scale + causal mask + online softmax (row = lg*8+r) --------
        float p0[8], p1[8];
#pragma unroll
        for (int r = 0; r < 8; ++r) {
            const int qi = q0 + lg * 8 + r;
            float s0 = s[0][r] * scale;
            float s1 = s[1][r] * scale;
            if (key0 + lh > qi)      s0 = -1e30f;
            if (key0 + 16 + lh > qi) s1 = -1e30f;
            float tmax = fmaxf(s0, s1);
#pragma unroll
            for (int m = 1; m < 16; m <<= 1)
                tmax = fmaxf(tmax, __shfl_xor(tmax, m, 32));
            const float mnew  = fmaxf(rowm[r], tmax);
            const float alpha = __expf(rowm[r] - mnew);
            const float e0 = __expf(s0 - mnew);
            const float e1 = __expf(s1 - mnew);
            float ps = e0 + e1;
#pragma unroll
            for (int m = 1; m < 16; m <<= 1)
                ps += __shfl_xor(ps, m, 32);
            rowl[r] = rowl[r] * alpha + ps;
            rowm[r] = mnew;
            p0[r] = e0; p1[r] = e1;
#pragma unroll
            for (int f = 0; f < 8; ++f) o[f][r] *= alpha;
        }

        // ---- re-layout P: D-fragment -> A-fragment via LDS --------------
        _Float16* pw = &plds[wave][0];
#pragma unroll
        for (int r = 0; r < 8; ++r) {
            const int row = lg * 8 + r;
            pw[row * 32 + lh]      = (_Float16)p0[r];
            pw[row * 32 + 16 + lh] = (_Float16)p1[r];
        }
        asm volatile("s_wait_dscnt 0x0" ::: "memory");
        v16h pf;
        {
            AFrag f;
            f.p.lo = *(const v8h*)(pw + lh * 32 + lg * 8);
            f.p.hi = *(const v8h*)(pw + lh * 32 + lg * 8 + 16);
            pf = f.v;
        }

        // ---- O += P * V (V tile from LDS, keys contiguous per lane) -----
#pragma unroll
        for (int f = 0; f < 8; ++f) {
            const _Float16* vp = &ldsV[cur][(f * 16 + lh) * 32 + lg * 8];
            AFrag g;
            g.p.lo = *(const v8h*)(vp);
            g.p.hi = *(const v8h*)(vp + 16);
            v16h vf = g.v;
            o[f] = __builtin_amdgcn_wmma_f32_16x16x32_f16(
                false, pf, false, vf, (short)0, o[f], false, false);
        }
    }

    // ---- normalize + store in [token, h*Dh + dim] layout -----------------
    const size_t outRow = (size_t)(b * T + q0 + lg * 8) * C + h * Dh;
#pragma unroll
    for (int f = 0; f < 8; ++f)
#pragma unroll
        for (int r = 0; r < 8; ++r) {
            float val = o[f][r] / fmaxf(rowl[r], 1e-20f);
            out[outRow + (size_t)r * C + f * 16 + lh] = (_Float16)val;
        }
}

// ---------------------------------------------------------------------------
extern "C" void kernel_launch(void* const* d_in, const int* in_sizes, int n_in,
                              void* d_out, int out_size, void* d_ws, size_t ws_size,
                              hipStream_t stream) {
    (void)in_sizes; (void)n_in; (void)out_size; (void)ws_size;

    constexpr int B = 4, T = 2048, C = 2048;
    constexpr int M  = B * T;
    constexpr int N3 = 3 * C;
    constexpr size_t nX   = (size_t)M * C;
    constexpr size_t nWa  = (size_t)N3 * C;
    constexpr size_t nWp  = (size_t)C * C;
    constexpr size_t nQKV = (size_t)M * N3;
    constexpr size_t nVT  = (size_t)B * 16 * 128 * T;

    const float* x  = (const float*)d_in[0];
    const float* Wa = (const float*)d_in[1];
    const float* Wp = (const float*)d_in[2];
    float* out = (float*)d_out;

    auto al = [](size_t v) { return (v + 255) & ~(size_t)255; };
    char* ws = (char*)d_ws;
    size_t off = 0;
    _Float16* xh   = (_Float16*)(ws + off); off = al(off + nX   * 2);
    _Float16* wah  = (_Float16*)(ws + off); off = al(off + nWa  * 2);
    _Float16* wph  = (_Float16*)(ws + off); off = al(off + nWp  * 2);
    _Float16* qkv  = (_Float16*)(ws + off); off = al(off + nQKV * 2);
    _Float16* vT   = (_Float16*)(ws + off); off = al(off + nVT  * 2);
    _Float16* attn = xh;   // reuse: x(f16) dead after QKV GEMM, same size

    cast_f32_f16<<<(int)(nX  / 1024), 256, 0, stream>>>(x,  xh,  (int)nX);
    cast_f32_f16<<<(int)(nWa / 1024), 256, 0, stream>>>(Wa, wah, (int)nWa);
    cast_f32_f16<<<(int)(nWp / 1024), 256, 0, stream>>>(Wp, wph, (int)nWp);

    gemm_wmma<_Float16><<<dim3(N3 / 128, M / 128), 256, 0, stream>>>(
        xh, wah, qkv, M, N3, C);

    transpose_v<<<(int)(nVT / 256), 256, 0, stream>>>(qkv, vT);

    flash_attn<<<dim3(T / 128, B * 16), 256, 0, stream>>>(qkv, vT, attn);

    gemm_wmma<float><<<dim3(C / 128, M / 128), 256, 0, stream>>>(
        attn, wph, out, M, C, C);
}
